// AttentionMechanism_82154134438381
// MI455X (gfx1250) — compile-verified, matrix operand order checked by reading, not run on hardware
//
#include <hip/hip_runtime.h>

// Problem constants
#define B_   32
#define C_   1024
#define N_   784     // 28*28, divisible by 16 (49 tiles)
#define NK_  800     // N padded to multiple of 32 for GEMM1 K-loop

typedef __attribute__((ext_vector_type(16))) __bf16 bf16x16;
typedef __attribute__((ext_vector_type(8)))  float  floatx8;

__device__ __forceinline__ unsigned short f2bf(float f) {
    union { float f; unsigned u; } v; v.f = f;
    unsigned r = v.u + 0x7FFFu + ((v.u >> 16) & 1u);   // round-to-nearest-even
    return (unsigned short)(r >> 16);
}

// ---------------------------------------------------------------------------
// Kernel 1: convert x (fp32, C x N row-major per batch) into
//   Xb  : bf16, C x NK_ (zero-padded cols 784..799)   -- A/B frags for GEMM1
//   XbT : bf16, N x C   (transpose)                   -- B frags for GEMM2
// LDS-tiled 32x32 transpose so both reads and writes are coalesced.
// ---------------------------------------------------------------------------
__global__ void prep_kernel(const float* __restrict__ x,
                            unsigned short* __restrict__ Xb,
                            unsigned short* __restrict__ XbT) {
    __shared__ float tile[32][33];
    const int b  = blockIdx.z;
    const int c0 = blockIdx.y * 32;
    const int n0 = blockIdx.x * 32;
    const float* xb = x + (size_t)b * C_ * N_;
#pragma unroll
    for (int ii = 0; ii < 4; ++ii) {
        const int i = ii * 8 + threadIdx.y;
        const int c = c0 + i;
        const int n = n0 + threadIdx.x;
        const float v = (n < N_) ? xb[(size_t)c * N_ + n] : 0.0f;
        tile[i][threadIdx.x] = v;
        Xb[((size_t)b * C_ + c) * NK_ + n] = f2bf(v);
    }
    __syncthreads();
#pragma unroll
    for (int ii = 0; ii < 4; ++ii) {
        const int nn = n0 + ii * 8 + threadIdx.y;
        const int cc = c0 + threadIdx.x;
        if (nn < N_)
            XbT[((size_t)b * N_ + nn) * C_ + cc] =
                f2bf(tile[threadIdx.x][ii * 8 + threadIdx.y]);
    }
}

// ---------------------------------------------------------------------------
// Kernel 2: S[b] = Xb[b] * Xb[b]^T, exploiting symmetry.
// One wave computes one 64x64 block-pair (i<=j) => 4x4 tiles, 16 WMMA / 8
// fragment loads per K=32 step.  Each tile is stored twice:
//   - normal orientation (scatter, scalar b32 stores)
//   - mirrored into the (j,i) block: contiguous over the 8 acc VGPRs ->
//     one 32-byte vector store per lane per tile.
// 136 pairs * 32 batches = 4352 waves = 544 blocks * 8 waves.
// ---------------------------------------------------------------------------
__global__ __launch_bounds__(256) void gemm1_kernel(
        const unsigned short* __restrict__ Xb, float* __restrict__ S) {
    const int lane = threadIdx.x & 31;
    const int w    = blockIdx.x * 8 + (threadIdx.x >> 5);
    const int b    = w / 136;           // 136 triangular pairs per batch
    int r          = w % 136;
    int i = 0;                           // decode (i <= j), wave-uniform SALU
    while (r >= (16 - i)) { r -= (16 - i); ++i; }
    const int j  = i + r;
    const int i0 = i * 64;
    const int j0 = j * 64;

    const unsigned short* Xbb = Xb + (size_t)b * C_ * NK_;
    const int rA   = lane & 15;
    const int koff = (lane >> 4) * 16;

    const unsigned short* pa = Xbb + (size_t)(i0 + rA) * NK_ + koff;
    const unsigned short* pb = Xbb + (size_t)(j0 + rA) * NK_ + koff;

    floatx8 acc[4][4];
#pragma unroll
    for (int t = 0; t < 4; ++t)
#pragma unroll
        for (int u = 0; u < 4; ++u) acc[t][u] = (floatx8)0.0f;

    for (int k = 0; k < NK_; k += 32) {
        bf16x16 af[4], bfr[4];
#pragma unroll
        for (int t = 0; t < 4; ++t)
            af[t] = *(const bf16x16*)(pa + (size_t)(16 * t) * NK_ + k);
#pragma unroll
        for (int u = 0; u < 4; ++u)
            bfr[u] = *(const bf16x16*)(pb + (size_t)(16 * u) * NK_ + k);
#pragma unroll
        for (int t = 0; t < 4; ++t)
#pragma unroll
            for (int u = 0; u < 4; ++u)
                acc[t][u] = __builtin_amdgcn_wmma_f32_16x16x32_bf16(
                    false, af[t], false, bfr[u], (short)0, acc[t][u], false, false);
    }

    float* Sb = S + (size_t)b * C_ * C_;
    const int mBase = i0 + (lane >> 4) * 8;  // M rows held by this lane
    const int nCol  = lane & 15;             // N column held by this lane
    // normal orientation: S[i-block][j-block] (scatter)
#pragma unroll
    for (int t = 0; t < 4; ++t)
#pragma unroll
        for (int u = 0; u < 4; ++u)
#pragma unroll
            for (int rr = 0; rr < 8; ++rr)
                Sb[(size_t)(mBase + 16 * t + rr) * C_ + (j0 + 16 * u + nCol)] =
                    acc[t][u][rr];
    // mirrored orientation: S[j-block][i-block] (contiguous 32B per lane)
#pragma unroll
    for (int t = 0; t < 4; ++t)
#pragma unroll
        for (int u = 0; u < 4; ++u)
            *(floatx8*)&Sb[(size_t)(j0 + 16 * u + nCol) * C_ +
                           (mBase + 16 * t)] = acc[t][u];
}

// ---------------------------------------------------------------------------
// Kernel 3: row softmax (fp32) of S, writing probabilities as bf16 aliased
// into the FRONT HALF of the same fp32 row (row byte stride stays C_*4).
// ---------------------------------------------------------------------------
__global__ __launch_bounds__(256) void softmax_kernel(float* __restrict__ S) {
    const int row = blockIdx.x;          // B_*C_ rows
    float* srow = S + (size_t)row * C_;
    unsigned short* prow = (unsigned short*)srow;   // bf16 alias, front half
    const int t = threadIdx.x;

    float v[4];
#pragma unroll
    for (int i = 0; i < 4; ++i) v[i] = srow[t + i * 256];

    __shared__ float red[8];
    // --- max ---
    float m = fmaxf(fmaxf(v[0], v[1]), fmaxf(v[2], v[3]));
#pragma unroll
    for (int off = 16; off > 0; off >>= 1) m = fmaxf(m, __shfl_xor(m, off, 32));
    if ((t & 31) == 0) red[t >> 5] = m;
    __syncthreads();
    float mAll = red[0];
#pragma unroll
    for (int i = 1; i < 8; ++i) mAll = fmaxf(mAll, red[i]);
    __syncthreads();
    // --- exp + sum ---
    float e[4], s = 0.0f;
#pragma unroll
    for (int i = 0; i < 4; ++i) { e[i] = __expf(v[i] - mAll); s += e[i]; }
#pragma unroll
    for (int off = 16; off > 0; off >>= 1) s += __shfl_xor(s, off, 32);
    if ((t & 31) == 0) red[t >> 5] = s;
    __syncthreads();
    float sAll = 0.0f;
#pragma unroll
    for (int i = 0; i < 8; ++i) sAll += red[i];
    const float inv = 1.0f / sAll;
    // all reads of srow happened before this barrier-protected write
#pragma unroll
    for (int i = 0; i < 4; ++i) prow[t + i * 256] = f2bf(e[i] * inv);
}

// ---------------------------------------------------------------------------
// Kernel 4: out = alpha * (P * X) + x.  P: bf16 rows aliased in S
// (row byte stride C_*4), B-fragments from XbT (contiguous 32B chunks).
// Wave tile 64x32 (8 WMMA / 6 loads per K=32 step) for the 768 aligned
// columns, 64x16 tail for columns 768..783.  25 column-jobs per row group.
// 16*25 jobs * 32 batches = 12800 waves = 1600 blocks * 8 waves.
// ---------------------------------------------------------------------------
template <int NT>
__device__ __forceinline__ void gemm2_tile(
        const char* __restrict__ Pb, const unsigned short* __restrict__ Tb,
        const float* __restrict__ xb, float* __restrict__ ob, float a,
        int i0, int n0, int lane) {
    const int rA   = lane & 15;
    const int koff = (lane >> 4) * 16;

    const char* pa = Pb + (size_t)(i0 + rA) * (C_ * 4) + (size_t)koff * 2;
    const unsigned short* pb = Tb + (size_t)(n0 + rA) * C_ + koff;

    floatx8 acc[4][NT];
#pragma unroll
    for (int t = 0; t < 4; ++t)
#pragma unroll
        for (int u = 0; u < NT; ++u) acc[t][u] = (floatx8)0.0f;

    for (int k = 0; k < C_; k += 32) {
        bf16x16 bfr[NT];
#pragma unroll
        for (int u = 0; u < NT; ++u)
            bfr[u] = *(const bf16x16*)(pb + (size_t)(16 * u) * C_ + k);
#pragma unroll
        for (int t = 0; t < 4; ++t) {
            const bf16x16 af = *(const bf16x16*)(
                pa + (size_t)(16 * t) * (C_ * 4) + (size_t)k * 2);
#pragma unroll
            for (int u = 0; u < NT; ++u)
                acc[t][u] = __builtin_amdgcn_wmma_f32_16x16x32_bf16(
                    false, af, false, bfr[u], (short)0, acc[t][u], false, false);
        }
    }

    const int mBase = i0 + (lane >> 4) * 8;
    const int col   = lane & 15;
#pragma unroll
    for (int t = 0; t < 4; ++t)
#pragma unroll
        for (int u = 0; u < NT; ++u)
#pragma unroll
            for (int rr = 0; rr < 8; ++rr) {
                const size_t idx =
                    (size_t)(mBase + 16 * t + rr) * N_ + (n0 + 16 * u + col);
                ob[idx] = a * acc[t][u][rr] + xb[idx];
            }
}

__global__ __launch_bounds__(256) void gemm2_kernel(
        const float* __restrict__ S, const unsigned short* __restrict__ XbT,
        const float* __restrict__ x, const float* __restrict__ alpha,
        float* __restrict__ out) {
    const int lane = threadIdx.x & 31;
    const int w    = blockIdx.x * 8 + (threadIdx.x >> 5);
    const int b    = w / 400;           // 16 row groups * 25 column jobs
    const int rem  = w % 400;
    const int i0   = (rem / 25) * 64;   // channel block
    const int jj   = rem % 25;          // column job

    const char* Pb = (const char*)(S + (size_t)b * C_ * C_);
    const unsigned short* Tb = XbT + (size_t)b * N_ * C_;
    const float* xb = x   + (size_t)b * C_ * N_;
    float*       ob = out + (size_t)b * C_ * N_;
    const float  a  = alpha[0];

    if (jj < 24)
        gemm2_tile<2>(Pb, Tb, xb, ob, a, i0, jj * 32, lane);   // 64x32
    else
        gemm2_tile<1>(Pb, Tb, xb, ob, a, i0, 768, lane);       // 64x16 tail
}

// ---------------------------------------------------------------------------
extern "C" void kernel_launch(void* const* d_in, const int* in_sizes, int n_in,
                              void* d_out, int out_size, void* d_ws, size_t ws_size,
                              hipStream_t stream) {
    const float* x     = (const float*)d_in[0];
    const float* alpha = (const float*)d_in[1];
    float* out = (float*)d_out;

    char* ws = (char*)d_ws;
    // Workspace layout (bytes):
    //   Xb  : 32*1024*800*2  = 52,428,800
    //   XbT : 32*784*1024*2  = 51,380,224
    //   S   : 32*1024*1024*4 = 134,217,728   (P bf16 aliased into row fronts)
    unsigned short* Xb  = (unsigned short*)(ws);
    unsigned short* XbT = (unsigned short*)(ws + 52428800ull);
    float*          S   = (float*)(ws + 103809024ull);

    dim3 pblk(32, 8);
    prep_kernel<<<dim3(25, 32, 32), pblk, 0, stream>>>(x, Xb, XbT);
    gemm1_kernel<<<544, 256, 0, stream>>>(Xb, S);       // 136 pairs/batch
    softmax_kernel<<<B_ * C_, 256, 0, stream>>>(S);
    gemm2_kernel<<<1600, 256, 0, stream>>>(S, XbT, x, alpha, out);
}
